// LSTMReturnModel_56427280335168
// MI455X (gfx1250) — matrix operand order repaired and checked by reading, not compiled, not run on hardware
//
#include <hip/hip_runtime.h>

#define B_TOT 1024
#define T_LEN 512
#define DIN   32
#define H     64
#define G     256   // 4*H

// element strides (padded for LDS bank spread, 16B-aligned rows)
#define LDW 264   // transposed weight rows [k][n], f16
#define LDZ 260   // gate matrix z rows [m][n], f32
#define LDH 72    // hidden state rows [m][k], f16
#define LDX 40    // x tile rows [m][k], f16

typedef __attribute__((ext_vector_type(16))) _Float16 v16h;
typedef __attribute__((ext_vector_type(8)))  _Float16 v8h_t;
typedef __attribute__((ext_vector_type(8)))  float    v8f;

__device__ __forceinline__ v16h make_v16(v8h_t lo, v8h_t hi) {
  return __builtin_shufflevector(lo, hi, 0,1,2,3,4,5,6,7,8,9,10,11,12,13,14,15);
}

// A fragment: 16x32 f16, M = lane&15, K halves per ISA 16-bit A layout.
// Two ds_load_b128 per lane.
__device__ __forceinline__ v16h load_a_frag(const _Float16* base, int ld, int kk, int lane) {
  int m = lane & 15;
  int koff = kk * 32 + ((lane & 16) ? 8 : 0);
  const _Float16* p = base + m * ld + koff;
  v8h_t lo = *(const v8h_t*)p;
  v8h_t hi = *(const v8h_t*)(p + 16);
  return make_v16(lo, hi);
}

// B fragment: 32x16 f16, N = ntile + (lane&15), K halves per ISA 16-bit layout (same as A).
// 16 scalar ds loads per lane.
__device__ __forceinline__ v16h load_b_frag(const _Float16* wt, int kbase, int ntile, int lane) {
  const _Float16* p = wt + (kbase + ((lane & 16) ? 8 : 0)) * LDW + ntile + (lane & 15);
  v16h r;
  #pragma unroll
  for (int s = 0; s < 8; ++s) { r[s] = p[s * LDW]; r[8 + s] = p[(16 + s) * LDW]; }
  return r;
}

__device__ __forceinline__ v8f wmma16(v16h a, v16h b, v8f c) {
  return __builtin_amdgcn_wmma_f32_16x16x32_f16(false, a, false, b, (short)0, c, false, false);
}

__device__ __forceinline__ float fsigmoid(float x) { return 1.f / (1.f + __expf(-x)); }
__device__ __forceinline__ float ftanh(float x)    { return 1.f - 2.f / (__expf(2.f * x) + 1.f); }

__global__ __launch_bounds__(256, 1)
void lstm2_fused_wmma(const float* __restrict__ x,
                      const float* __restrict__ Wih0, const float* __restrict__ Whh0,
                      const float* __restrict__ bih0, const float* __restrict__ bhh0,
                      const float* __restrict__ Wih1, const float* __restrict__ Whh1,
                      const float* __restrict__ bih1, const float* __restrict__ bhh1,
                      const float* __restrict__ W1,   const float* __restrict__ b1,
                      const float* __restrict__ W2,   const float* __restrict__ b2,
                      float* __restrict__ out)
{
  extern __shared__ char smem[];
  float*    zls  = (float*)smem;                               // 16*LDZ f32
  _Float16* wih0 = (_Float16*)(smem + 16 * LDZ * sizeof(float));
  _Float16* whh0 = wih0 + 32 * LDW;
  _Float16* wih1 = whh0 + 64 * LDW;
  _Float16* whh1 = wih1 + 64 * LDW;
  _Float16* hA   = whh1 + 64 * LDW;    // layer0 hidden state (16xH f16)
  _Float16* hB   = hA + 16 * LDH;      // layer1 hidden state
  _Float16* xs   = hB + 16 * LDH;      // x_t tile (16xDIN f16)

  const int tid  = threadIdx.x;
  const int lane = tid & 31;
  const int wave = tid >> 5;
  const int wg   = blockIdx.x;

  // ---- one-time: transpose + f16-convert weights into LDS ----
  for (int idx = tid; idx < G * DIN; idx += 256) {
    int n = idx >> 5, k = idx & 31;
    wih0[k * LDW + n] = (_Float16)Wih0[idx];
  }
  for (int idx = tid; idx < G * H; idx += 256) {
    int n = idx >> 6, k = idx & 63;
    whh0[k * LDW + n] = (_Float16)Whh0[idx];
    wih1[k * LDW + n] = (_Float16)Wih1[idx];
    whh1[k * LDW + n] = (_Float16)Whh1[idx];
  }
  for (int p = tid; p < 16 * LDH; p += 256) { hA[p] = (_Float16)0.f; hB[p] = (_Float16)0.f; }

  // per-wave N tiles and per-lane fused biases (folded into WMMA C init)
  const int nt0 = wave * 32;
  const int nt1 = nt0 + 16;
  const int nc0 = nt0 + (lane & 15);
  const int nc1 = nc0 + 16;
  const float bia0_0 = bih0[nc0] + bhh0[nc0];
  const float bia0_1 = bih0[nc1] + bhh0[nc1];
  const float bia1_0 = bih1[nc0] + bhh1[nc0];
  const float bia1_1 = bih1[nc1] + bhh1[nc1];

  float cA[4] = {0.f, 0.f, 0.f, 0.f};
  float cB[4] = {0.f, 0.f, 0.f, 0.f};

  const int xm = tid >> 4;
  const int xk = (tid & 15) * 2;
  const float* xrow = x + ((size_t)(wg * 16 + xm) * T_LEN) * DIN + xk;

  const int zr  = (lane >> 4) * 8;      // C/D layout: M = vgpr + 8*(lane>=16)
  const int zc0 = nt0 + (lane & 15);
  const int zc1 = zc0 + 16;

  __syncthreads();

  for (int t = 0; t < T_LEN; ++t) {
    // stage x_t -> LDS f16 (coalesced 8B per thread)
    float2 xv = *(const float2*)(xrow + (size_t)t * DIN);
    xs[xm * LDX + xk]     = (_Float16)xv.x;
    xs[xm * LDX + xk + 1] = (_Float16)xv.y;
    __syncthreads();

    // ---- layer0: z = x@Wih0^T + hA@Whh0^T + bias ----
    v8f a0, a1;
    #pragma unroll
    for (int i = 0; i < 8; ++i) { a0[i] = bia0_0; a1[i] = bia0_1; }
    {
      v16h af = load_a_frag(xs, LDX, 0, lane);
      a0 = wmma16(af, load_b_frag(wih0, 0, nt0, lane), a0);
      a1 = wmma16(af, load_b_frag(wih0, 0, nt1, lane), a1);
      af = load_a_frag(hA, LDH, 0, lane);
      a0 = wmma16(af, load_b_frag(whh0, 0, nt0, lane), a0);
      a1 = wmma16(af, load_b_frag(whh0, 0, nt1, lane), a1);
      af = load_a_frag(hA, LDH, 1, lane);
      a0 = wmma16(af, load_b_frag(whh0, 32, nt0, lane), a0);
      a1 = wmma16(af, load_b_frag(whh0, 32, nt1, lane), a1);
    }
    #pragma unroll
    for (int v = 0; v < 8; ++v) {
      zls[(zr + v) * LDZ + zc0] = a0[v];
      zls[(zr + v) * LDZ + zc1] = a1[v];
    }
    __syncthreads();

    // ---- layer0 gates / state update (c in regs, h -> LDS f16) ----
    #pragma unroll
    for (int r = 0; r < 4; ++r) {
      int idx = tid + 256 * r;
      int m = idx >> 6, j = idx & 63;
      const float* zp = zls + m * LDZ;
      float ig = fsigmoid(zp[j]);
      float fg = fsigmoid(zp[64 + j]);
      float gg = ftanh(zp[128 + j]);
      float og = fsigmoid(zp[192 + j]);
      cA[r] = fg * cA[r] + ig * gg;
      hA[m * LDH + j] = (_Float16)(og * ftanh(cA[r]));
    }
    __syncthreads();

    // ---- layer1: z = hA@Wih1^T + hB@Whh1^T + bias ----
    #pragma unroll
    for (int i = 0; i < 8; ++i) { a0[i] = bia1_0; a1[i] = bia1_1; }
    {
      v16h af = load_a_frag(hA, LDH, 0, lane);
      a0 = wmma16(af, load_b_frag(wih1, 0, nt0, lane), a0);
      a1 = wmma16(af, load_b_frag(wih1, 0, nt1, lane), a1);
      af = load_a_frag(hA, LDH, 1, lane);
      a0 = wmma16(af, load_b_frag(wih1, 32, nt0, lane), a0);
      a1 = wmma16(af, load_b_frag(wih1, 32, nt1, lane), a1);
      af = load_a_frag(hB, LDH, 0, lane);
      a0 = wmma16(af, load_b_frag(whh1, 0, nt0, lane), a0);
      a1 = wmma16(af, load_b_frag(whh1, 0, nt1, lane), a1);
      af = load_a_frag(hB, LDH, 1, lane);
      a0 = wmma16(af, load_b_frag(whh1, 32, nt0, lane), a0);
      a1 = wmma16(af, load_b_frag(whh1, 32, nt1, lane), a1);
    }
    #pragma unroll
    for (int v = 0; v < 8; ++v) {
      zls[(zr + v) * LDZ + zc0] = a0[v];
      zls[(zr + v) * LDZ + zc1] = a1[v];
    }
    __syncthreads();

    // ---- layer1 gates / state update ----
    #pragma unroll
    for (int r = 0; r < 4; ++r) {
      int idx = tid + 256 * r;
      int m = idx >> 6, j = idx & 63;
      const float* zp = zls + m * LDZ;
      float ig = fsigmoid(zp[j]);
      float fg = fsigmoid(zp[64 + j]);
      float gg = ftanh(zp[128 + j]);
      float og = fsigmoid(zp[192 + j]);
      cB[r] = fg * cB[r] + ig * gg;
      hB[m * LDH + j] = (_Float16)(og * ftanh(cB[r]));
    }
    __syncthreads();
  }

  // ---- head: hidden = relu(hB_last @ W1^T + b1); out = hidden @ W2^T + b2 ----
  float* hid = zls;  // reuse z LDS (16x32 f32)
  for (int p = tid; p < 16 * 32; p += 256) {
    int m = p >> 5, j2 = p & 31;
    float acc = b1[j2];
    const float* w1r = W1 + j2 * H;
    #pragma unroll
    for (int k = 0; k < H; ++k) acc += (float)hB[m * LDH + k] * w1r[k];
    hid[m * 32 + j2] = fmaxf(acc, 0.f);
  }
  __syncthreads();
  if (tid < 16) {
    float acc = b2[0];
    #pragma unroll
    for (int j2 = 0; j2 < 32; ++j2) acc += hid[tid * 32 + j2] * W2[j2];
    out[wg * 16 + tid] = acc;
  }
}

extern "C" void kernel_launch(void* const* d_in, const int* in_sizes, int n_in,
                              void* d_out, int out_size, void* d_ws, size_t ws_size,
                              hipStream_t stream) {
  (void)in_sizes; (void)n_in; (void)d_ws; (void)ws_size; (void)out_size;
  const float* x    = (const float*)d_in[0];
  const float* Wih0 = (const float*)d_in[1];
  const float* Whh0 = (const float*)d_in[2];
  const float* bih0 = (const float*)d_in[3];
  const float* bhh0 = (const float*)d_in[4];
  const float* Wih1 = (const float*)d_in[5];
  const float* Whh1 = (const float*)d_in[6];
  const float* bih1 = (const float*)d_in[7];
  const float* bhh1 = (const float*)d_in[8];
  const float* W1   = (const float*)d_in[9];
  const float* b1   = (const float*)d_in[10];
  const float* W2   = (const float*)d_in[11];
  const float* b2   = (const float*)d_in[12];
  float* out = (float*)d_out;

  size_t shmem = (size_t)16 * LDZ * sizeof(float)
               + (size_t)(32 * LDW + 3 * 64 * LDW + 2 * 16 * LDH + 16 * LDX) * 2u;

  lstm2_fused_wmma<<<B_TOT / 16, 256, shmem, stream>>>(
      x, Wih0, Whh0, bih0, bhh0, Wih1, Whh1, bih1, bhh1, W1, b1, W2, b2, out);
}
